// NeuralModel_59167469469858
// MI455X (gfx1250) — matrix operand order repaired and checked by reading, compiled, not run on hardware
//
#include <hip/hip_runtime.h>

typedef __attribute__((ext_vector_type(16))) _Float16 v16h;
typedef __attribute__((ext_vector_type(8)))  _Float16 v8h;
typedef __attribute__((ext_vector_type(8)))  float    v8f;
typedef __attribute__((ext_vector_type(4)))  int      v4i;

#define NVARS  40000
#define NEDGE  200000
#define NCONF_ 100000
#define FEATD  128
#define TMAX   3
#define EPS_   1e-5f

#if defined(__gfx1250__) && __has_builtin(__builtin_amdgcn_global_load_async_to_lds_b128)
#define ASYNC_OK 1
typedef __attribute__((address_space(3))) v4i lds_v4i;
typedef __attribute__((address_space(1))) v4i glb_v4i;
#else
#define ASYNC_OK 0
#endif

// ---------------- workspace layout (bytes) ----------------
constexpr size_t OFF_STATEH = 0;                                       // NVARS*FEATD f16
constexpr size_t OFF_W1F    = OFF_STATEH + (size_t)NVARS*FEATD*2;      // 8*8*32*16 f16 frags
constexpr size_t OFF_W2F    = OFF_W1F + (size_t)8*8*32*16*2;           // 4*8*32*16 f16 frags
constexpr size_t OFF_AGG    = OFF_W2F + (size_t)4*8*32*16*2;           // NVARS*FEATD f32
constexpr size_t OFF_BNS    = OFF_AGG + (size_t)NVARS*FEATD*4;         // 256 f32 (sum, sumsq)
constexpr size_t OFF_LOG    = OFF_BNS + 256*4;                         // NVARS*TMAX*4 f32
constexpr size_t OFF_ASG    = OFF_LOG + (size_t)NVARS*TMAX*4*4;        // NVARS*TMAX i32
constexpr size_t OFF_SCL    = OFF_ASG + (size_t)NVARS*TMAX*4;          // 3 f32

// ---------------- output layout (float elements) ----------------
constexpr size_t O_PHI   = 2;                                          // phi (N,T,D)
constexpr size_t O_CONF  = O_PHI + (size_t)NVARS*TMAX*4;               // edge_conflicts (E,T)
constexpr size_t O_ASG   = O_CONF + (size_t)NEDGE*TMAX;                // assignment[:, -1]
constexpr size_t O_NCONF = O_ASG + (size_t)NVARS;                      // n_conf

union H16 { v16h v; v8h h[2]; };

// ------------------------------------------------------------------
__global__ __launch_bounds__(256) void k_cvt_state(const float* __restrict__ s0,
                                                   _Float16* __restrict__ sh)
{
    size_t i = (size_t)blockIdx.x*blockDim.x + threadIdx.x;
    for (size_t k = i; k < (size_t)NVARS*FEATD; k += (size_t)gridDim.x*blockDim.x)
        sh[k] = (_Float16)s0[k];
}

// Pre-swizzle W1/W2 into WMMA B-fragment order:
// frag element index = ((kb*8+nt)*32 + lane)*16 + h,  value = W[kb*32 + (h<8 ? (lane>=16?8:0) + h : 16 + (lane>=16?8:0) + h-8)][nt*16 + lane%16]
__global__ __launch_bounds__(256) void k_cvt_w(const float* __restrict__ W1,
                                               const float* __restrict__ W2,
                                               _Float16* __restrict__ w1f,
                                               _Float16* __restrict__ w2f,
                                               float* __restrict__ scal)
{
    int gid = blockIdx.x*blockDim.x + threadIdx.x;
    if (gid < 3) scal[gid] = 0.f;
    const int N1 = 8*8*32*16;
    const int N2 = 4*8*32*16;
    if (gid < N1) {
        int p = gid;
        int h = p & 15, lane = (p>>4)&31, nt = (p>>9)&7, kb = p>>12;
        int K = kb*32 + ((lane & 16) ? 8 : 0) + (h < 8 ? h : 16 + h - 8);
        int n = nt*16 + (lane & 15);
        w1f[p] = (_Float16)W1[K*FEATD + n];
    } else if (gid < N1 + N2) {
        int p = gid - N1;
        int h = p & 15, lane = (p>>4)&31, nt = (p>>9)&7, kb = p>>12;
        int K = kb*32 + ((lane & 16) ? 8 : 0) + (h < 8 ? h : 16 + h - 8);
        int n = nt*16 + (lane & 15);
        w2f[p] = (_Float16)W2[K*FEATD + n];
    }
}

__global__ __launch_bounds__(256) void k_zero(float* __restrict__ agg, float* __restrict__ bns)
{
    size_t i = (size_t)blockIdx.x*blockDim.x + threadIdx.x;
    for (size_t k = i; k < (size_t)NVARS*FEATD; k += (size_t)gridDim.x*blockDim.x) agg[k] = 0.f;
    if (i < 256) bns[i] = 0.f;
}

// ------------------------------------------------------------------
__device__ __forceinline__ void gather_fence(bool hasNext)
{
#if ASYNC_OK
    if (hasNext) {
#if __has_builtin(__builtin_amdgcn_s_wait_asynccnt)
        __builtin_amdgcn_s_wait_asynccnt(16);   // current tile's 16 ops done (in-order)
#else
        asm volatile("s_wait_asynccnt 16" ::: "memory");
#endif
    } else {
#if __has_builtin(__builtin_amdgcn_s_wait_asynccnt)
        __builtin_amdgcn_s_wait_asynccnt(0);
#else
        asm volatile("s_wait_asynccnt 0" ::: "memory");
#endif
    }
#else
    (void)hasNext;
    asm volatile("s_wait_dscnt 0" ::: "memory");
#endif
    asm volatile("" ::: "memory");
}

// Fused edge MLP: async double-buffered gather -> GEMM1(WMMA) -> ReLU
//                 -> GEMM2(WMMA) -> ReLU -> LayerNorm -> atomic scatter-add
// block = 64 threads = 2 waves, each wave owns one 16-row message tile (2 buffers).
__global__ __launch_bounds__(64) void k_edge_mlp(
    const _Float16* __restrict__ stateH,
    const _Float16* __restrict__ w1f,
    const _Float16* __restrict__ w2f,
    const float* __restrict__ b1,
    const float* __restrict__ b2,
    const float* __restrict__ lng,
    const float* __restrict__ lnb,
    const int* __restrict__ idxL,
    const int* __restrict__ idxR,
    float* __restrict__ agg)
{
    __shared__ _Float16 sMsg[2][2][16*256];  // [wave][buf] msg tile; buf reused as f32 h2
    __shared__ _Float16 sH[2][16*128];       // [wave] hidden tile; reused as f32 LN stats

    const int lane  = threadIdx.x & 31;
    const int w     = threadIdx.x >> 5;
    const int mrow  = lane & 15;
    const int abase = (lane < 16) ? 0 : 8;   // A-fragment K sub-base (ISA 16-bit A layout)
    const int ncol  = lane & 15;             // C-layout column
    const int mofs  = (lane >> 4) * 8;       // C-layout row offset

    float bias1[8], bias2[8];
#pragma unroll
    for (int nt = 0; nt < 8; ++nt) { bias1[nt] = b1[nt*16+ncol]; bias2[nt] = b2[nt*16+ncol]; }
    float gv[4], bv[4];
#pragma unroll
    for (int j = 0; j < 4; ++j) { gv[j] = lng[lane+32*j]; bv[j] = lnb[lane+32*j]; }

    const v8f vzero = {0.f,0.f,0.f,0.f,0.f,0.f,0.f,0.f};
    const int nTiles = (2*NEDGE)/16;         // 25000, exact
    const int stride = gridDim.x*2;

    // issue one 16-row (16x256 f16) gather into sMsg[w][bufG]
    auto issue_gather = [&](int tileG, int bufG) {
        const int m = lane >> 1, seg = lane & 1;
        const int r = tileG*16 + m;
        const int e = (r < NEDGE) ? r : (r - NEDGE);
        const int srcRow = (r < NEDGE) ? (seg ? idxL[e] : idxR[e])
                                       : (seg ? idxR[e] : idxL[e]);
        // prefetch edge indices two strides ahead
        const int rp = r + 2*stride*16;
        if (rp < NEDGE) { __builtin_prefetch(idxL + rp, 0, 0); __builtin_prefetch(idxR + rp, 0, 0); }
#if ASYNC_OK
        glb_v4i* src = (glb_v4i*)(stateH + (size_t)srcRow*FEATD);
        lds_v4i* dst = (lds_v4i*)(&sMsg[w][bufG][m*256 + seg*128]);
#pragma unroll
        for (int i = 0; i < 16; ++i)
            __builtin_amdgcn_global_load_async_to_lds_b128(src + i, dst + i, 0, 0);
#else
        const uint4* src = (const uint4*)(stateH + (size_t)srcRow*FEATD);
        uint4* dst = (uint4*)(&sMsg[w][bufG][m*256 + seg*128]);
#pragma unroll
        for (int i = 0; i < 16; ++i) dst[i] = src[i];
#endif
    };

    int tile = blockIdx.x*2 + w;
    int buf  = 0;
    if (tile < nTiles) issue_gather(tile, buf);

    for (; tile < nTiles; tile += stride) {
        const int  nextTile = tile + stride;
        const bool hasNext  = nextTile < nTiles;
        // h2 (f32 view of the other buffer) reads must land before async rewrites it
        asm volatile("s_wait_dscnt 0" ::: "memory");
        if (hasNext) issue_gather(nextTile, buf ^ 1);
        gather_fence(hasNext);

        const int R = tile*16;
        _Float16* msg = &sMsg[w][buf][0];

        // ---- GEMM1: (16x256) @ (256x128), K-steps of 32 ----
        v8f acc[8];
#pragma unroll
        for (int i = 0; i < 8; ++i) acc[i] = vzero;
#pragma unroll
        for (int kb = 0; kb < 8; ++kb) {
            H16 a;
            a.h[0] = *(const v8h*)(&msg[mrow*256 + kb*32 + abase]);
            a.h[1] = *(const v8h*)(&msg[mrow*256 + kb*32 + 16 + abase]);
#pragma unroll
            for (int nt = 0; nt < 8; ++nt) {
                H16 b;
                const _Float16* bp = w1f + ((size_t)(kb*8+nt)*32 + lane)*16;
                b.h[0] = *(const v8h*)bp;
                b.h[1] = *(const v8h*)(bp + 8);
                acc[nt] = __builtin_amdgcn_wmma_f32_16x16x32_f16(
                    false, a.v, false, b.v, (short)0, acc[nt], false, false);
            }
        }
        // ---- bias + ReLU -> f16 hidden tile ----
#pragma unroll
        for (int nt = 0; nt < 8; ++nt) {
#pragma unroll
            for (int v = 0; v < 8; ++v) {
                float x = acc[nt][v] + bias1[nt];
                x = x > 0.f ? x : 0.f;
                sH[w][(mofs+v)*128 + nt*16 + ncol] = (_Float16)x;
            }
        }
        asm volatile("s_wait_dscnt 0" ::: "memory");

        // ---- GEMM2: (16x128) @ (128x128) ----
        v8f acc2[8];
#pragma unroll
        for (int i = 0; i < 8; ++i) acc2[i] = vzero;
#pragma unroll
        for (int kb = 0; kb < 4; ++kb) {
            H16 a;
            a.h[0] = *(const v8h*)(&sH[w][mrow*128 + kb*32 + abase]);
            a.h[1] = *(const v8h*)(&sH[w][mrow*128 + kb*32 + 16 + abase]);
#pragma unroll
            for (int nt = 0; nt < 8; ++nt) {
                H16 b;
                const _Float16* bp = w2f + ((size_t)(kb*8+nt)*32 + lane)*16;
                b.h[0] = *(const v8h*)bp;
                b.h[1] = *(const v8h*)(bp + 8);
                acc2[nt] = __builtin_amdgcn_wmma_f32_16x16x32_f16(
                    false, a.v, false, b.v, (short)0, acc2[nt], false, false);
            }
        }
        // ---- bias + ReLU -> f32 tile (reuse current msg buffer) ----
        float* h2 = (float*)msg;
#pragma unroll
        for (int nt = 0; nt < 8; ++nt) {
#pragma unroll
            for (int v = 0; v < 8; ++v) {
                float x = acc2[nt][v] + bias2[nt];
                h2[(mofs+v)*128 + nt*16 + ncol] = x > 0.f ? x : 0.f;
            }
        }
        asm volatile("s_wait_dscnt 0" ::: "memory");

        // ---- LayerNorm stats per row (one row per lane in lanes 0..15) ----
        float* stats = (float*)&sH[w][0];
        if (lane < 16) {
            float s = 0.f, ss = 0.f;
            for (int i = 0; i < 128; ++i) { float x = h2[lane*128+i]; s += x; ss += x*x; }
            float mu  = s * (1.f/128.f);
            float var = ss * (1.f/128.f) - mu*mu;
            stats[lane]      = mu;
            stats[16 + lane] = rsqrtf(var + EPS_);
        }
        asm volatile("s_wait_dscnt 0" ::: "memory");

        // ---- normalize + scatter-add into agg ----
        for (int m = 0; m < 16; ++m) {
            const int r   = R + m;
            const int tgt = (r < NEDGE) ? idxL[r] : idxR[r - NEDGE];
            const float mu = stats[m], rs = stats[16+m];
            float* arow = agg + (size_t)tgt*FEATD;
#pragma unroll
            for (int j = 0; j < 4; ++j) {
                float x = h2[m*128 + lane + 32*j];
                atomicAdd(arow + lane + 32*j, gv[j]*(x - mu)*rs + bv[j]);
            }
        }
        buf ^= 1;
    }
}

// ------------------------------------------------------------------
__global__ __launch_bounds__(256) void k_bn_stats(const float* __restrict__ agg,
                                                  const float* __restrict__ degs,
                                                  float* __restrict__ bns)
{
    __shared__ float ss[256], sq[256];
    const int f    = threadIdx.x & 127;
    const int half = threadIdx.x >> 7;
    float s = 0.f, q = 0.f;
    for (int n = blockIdx.x*2 + half; n < NVARS; n += gridDim.x*2) {
        float x = agg[(size_t)n*FEATD + f] / degs[n];
        s += x; q += x*x;
    }
    ss[threadIdx.x] = s; sq[threadIdx.x] = q;
    __syncthreads();
    if (half == 0) {
        atomicAdd(&bns[f],       ss[f] + ss[f+128]);
        atomicAdd(&bns[128 + f], sq[f] + sq[f+128]);
    }
}

__global__ __launch_bounds__(128) void k_bn_apply(const float* __restrict__ agg,
                                                  const float* __restrict__ degs,
                                                  const float* __restrict__ bns,
                                                  const float* __restrict__ bng,
                                                  const float* __restrict__ bnb,
                                                  const float* __restrict__ Wd,
                                                  const float* __restrict__ bd,
                                                  _Float16* __restrict__ stateH,
                                                  float* __restrict__ logits,
                                                  int t)
{
    __shared__ float red[128*4];
    const int f = threadIdx.x;
    const float mu  = bns[f] * (1.f/(float)NVARS);
    const float var = bns[128+f] * (1.f/(float)NVARS) - mu*mu;
    const float rs  = rsqrtf(var + EPS_);
    const float g = bng[f], b = bnb[f];
    const float w0 = Wd[f*4+0], w1 = Wd[f*4+1], w2 = Wd[f*4+2], w3 = Wd[f*4+3];

    for (int n = blockIdx.x; n < NVARS; n += gridDim.x) {
        float x = agg[(size_t)n*FEATD + f] / degs[n];
        float s = g*(x - mu)*rs + b;
        stateH[(size_t)n*FEATD + f] = (_Float16)s;
        red[f*4+0] = s*w0; red[f*4+1] = s*w1; red[f*4+2] = s*w2; red[f*4+3] = s*w3;
        __syncthreads();
        for (int off = 64; off > 0; off >>= 1) {
            if (f < off) {
                red[f*4+0] += red[(f+off)*4+0];
                red[f*4+1] += red[(f+off)*4+1];
                red[f*4+2] += red[(f+off)*4+2];
                red[f*4+3] += red[(f+off)*4+3];
            }
            __syncthreads();
        }
        if (f < 4) logits[(size_t)n*(TMAX*4) + t*4 + f] = red[f] + bd[f];
        __syncthreads();
    }
}

// ------------------------------------------------------------------
__global__ __launch_bounds__(256) void k_softmax(const float* __restrict__ logits,
                                                 float* __restrict__ out,
                                                 int* __restrict__ assign)
{
    const int n = blockIdx.x*blockDim.x + threadIdx.x;
    if (n >= NVARS) return;
    int alast = 0;
    for (int t = 0; t < TMAX; ++t) {
        float l[4];
#pragma unroll
        for (int d = 0; d < 4; ++d) l[d] = logits[(size_t)n*(TMAX*4) + t*4 + d];
        float m = l[0]; int am = 0;
#pragma unroll
        for (int d = 1; d < 4; ++d) if (l[d] > m) { m = l[d]; am = d; }
        float e[4], s = 0.f;
#pragma unroll
        for (int d = 0; d < 4; ++d) { e[d] = expf(l[d] - m); s += e[d]; }
        float inv = 1.f / s;
#pragma unroll
        for (int d = 0; d < 4; ++d) out[O_PHI + (size_t)n*(TMAX*4) + t*4 + d] = e[d]*inv;
        assign[n*TMAX + t] = am;
        if (t == TMAX-1) alast = am;
    }
    out[O_ASG + n] = (float)alast;
}

__global__ __launch_bounds__(256) void k_edge_loss(float* __restrict__ out,
                                                   const int* __restrict__ assign,
                                                   const float* __restrict__ relation,
                                                   const int* __restrict__ idxL,
                                                   const int* __restrict__ idxR,
                                                   float* __restrict__ scal)
{
    __shared__ float sl[256], sc[256];
    float rel[16];
#pragma unroll
    for (int i = 0; i < 16; ++i) rel[i] = relation[i];
    float lacc = 0.f, cacc = 0.f;
    for (int e = blockIdx.x*blockDim.x + threadIdx.x; e < NEDGE; e += gridDim.x*blockDim.x) {
        const int il = idxL[e], ir = idxR[e];
        float pl[12], pr[12];
#pragma unroll
        for (int k = 0; k < 12; ++k) { pl[k] = out[O_PHI + (size_t)il*12 + k];
                                       pr[k] = out[O_PHI + (size_t)ir*12 + k]; }
        for (int t = 0; t < TMAX; ++t) {
            float s = 0.f;
#pragma unroll
            for (int i = 0; i < 4; ++i)
#pragma unroll
                for (int j = 0; j < 4; ++j)
                    s += pl[t*4+i] * rel[i*4+j] * pr[t*4+j];
            lacc += -logf(s);
            int al = assign[il*TMAX + t], ar = assign[ir*TMAX + t];
            float c = 1.f - rel[al*4 + ar];
            out[O_CONF + (size_t)e*TMAX + t] = c;
            if (t == TMAX-1) cacc += c;
        }
    }
    sl[threadIdx.x] = lacc; sc[threadIdx.x] = cacc;
    __syncthreads();
    for (int off = 128; off > 0; off >>= 1) {
        if (threadIdx.x < off) { sl[threadIdx.x] += sl[threadIdx.x+off];
                                 sc[threadIdx.x] += sc[threadIdx.x+off]; }
        __syncthreads();
    }
    if (threadIdx.x == 0) { atomicAdd(&scal[0], sl[0]); atomicAdd(&scal[2], sc[0]); }
}

__global__ __launch_bounds__(256) void k_conf_loss(const float* __restrict__ out,
                                                   const float* __restrict__ cmat,
                                                   const int* __restrict__ idxLp,
                                                   const int* __restrict__ idxRp,
                                                   float* __restrict__ scal)
{
    __shared__ float sl[256];
    float mat[16];
#pragma unroll
    for (int i = 0; i < 16; ++i) mat[i] = cmat[i];
    float lacc = 0.f;
    for (int e = blockIdx.x*blockDim.x + threadIdx.x; e < NCONF_; e += gridDim.x*blockDim.x) {
        const int il = idxLp[e], ir = idxRp[e];
        float pl[12], pr[12];
#pragma unroll
        for (int k = 0; k < 12; ++k) { pl[k] = out[O_PHI + (size_t)il*12 + k];
                                       pr[k] = out[O_PHI + (size_t)ir*12 + k]; }
        for (int t = 0; t < TMAX; ++t) {
            float s = 0.f;
#pragma unroll
            for (int i = 0; i < 4; ++i)
#pragma unroll
                for (int j = 0; j < 4; ++j)
                    s += pl[t*4+i] * mat[i*4+j] * pr[t*4+j];
            lacc += -logf(s);
        }
    }
    sl[threadIdx.x] = lacc;
    __syncthreads();
    for (int off = 128; off > 0; off >>= 1) {
        if (threadIdx.x < off) sl[threadIdx.x] += sl[threadIdx.x+off];
        __syncthreads();
    }
    if (threadIdx.x == 0) atomicAdd(&scal[1], sl[0]);
}

__global__ void k_final(const float* __restrict__ scal, float* __restrict__ out)
{
    if (threadIdx.x == 0 && blockIdx.x == 0) {
        float loss = scal[0]/(float)NEDGE + scal[1]/(10.f*(float)NCONF_);
        out[0] = loss;
        out[1] = scal[2]/(float)NEDGE;
        out[O_NCONF] = scal[2];
    }
}

// ------------------------------------------------------------------
extern "C" void kernel_launch(void* const* d_in, const int* in_sizes, int n_in,
                              void* d_out, int out_size, void* d_ws, size_t ws_size,
                              hipStream_t stream)
{
    (void)in_sizes; (void)n_in; (void)out_size; (void)ws_size;
    const float* state0   = (const float*)d_in[0];
    const float* W1       = (const float*)d_in[1];
    const float* b1       = (const float*)d_in[2];
    const float* W2       = (const float*)d_in[3];
    const float* b2       = (const float*)d_in[4];
    const float* ln_g     = (const float*)d_in[5];
    const float* ln_b     = (const float*)d_in[6];
    const float* bn_g     = (const float*)d_in[7];
    const float* bn_b     = (const float*)d_in[8];
    const float* Wd       = (const float*)d_in[9];
    const float* bd       = (const float*)d_in[10];
    const float* relation = (const float*)d_in[11];
    const float* confmat  = (const float*)d_in[12];
    const float* degrees  = (const float*)d_in[13];
    const int*   idxL     = (const int*)d_in[14];
    const int*   idxR     = (const int*)d_in[15];
    const int*   idxLp    = (const int*)d_in[16];
    const int*   idxRp    = (const int*)d_in[17];
    // d_in[18] = iterations (== T_MAX == 3, hardcoded)

    char* ws = (char*)d_ws;
    _Float16* stateH = (_Float16*)(ws + OFF_STATEH);
    _Float16* w1f    = (_Float16*)(ws + OFF_W1F);
    _Float16* w2f    = (_Float16*)(ws + OFF_W2F);
    float*    agg    = (float*)(ws + OFF_AGG);
    float*    bns    = (float*)(ws + OFF_BNS);
    float*    logits = (float*)(ws + OFF_LOG);
    int*      assign = (int*)(ws + OFF_ASG);
    float*    scal   = (float*)(ws + OFF_SCL);
    float*    out    = (float*)d_out;

    k_cvt_state<<<(NVARS*FEATD)/256, 256, 0, stream>>>(state0, stateH);
    k_cvt_w<<<192, 256, 0, stream>>>(W1, W2, w1f, w2f, scal);

    for (int t = 0; t < TMAX; ++t) {
        k_zero<<<2048, 256, 0, stream>>>(agg, bns);
        k_edge_mlp<<<2500, 64, 0, stream>>>(stateH, w1f, w2f, b1, b2, ln_g, ln_b,
                                            idxL, idxR, agg);
        k_bn_stats<<<512, 256, 0, stream>>>(agg, degrees, bns);
        k_bn_apply<<<2048, 128, 0, stream>>>(agg, degrees, bns, bn_g, bn_b, Wd, bd,
                                             stateH, logits, t);
    }

    k_softmax<<<(NVARS + 255)/256, 256, 0, stream>>>(logits, out, assign);
    k_edge_loss<<<1024, 256, 0, stream>>>(out, assign, relation, idxL, idxR, scal);
    k_conf_loss<<<512, 256, 0, stream>>>(out, confmat, idxLp, idxRp, scal);
    k_final<<<1, 1, 0, stream>>>(scal, out);
}
